// SpeechToTextRNN_38397007626832
// MI455X (gfx1250) — matrix operand order repaired and checked by reading, not compiled, not run on hardware
//
#include <hip/hip_runtime.h>

// ---------------------------------------------------------------------------
// 2-layer GRU + FC for MI455X (gfx1250), bf16 WMMA with fp32 accumulate.
// B=256, T=2000, F=80, H=64, O=29 (G=3H=192).
// 16 persistent WGs (16 batch rows each) scan T; 12 waves/WG, one 16-col gate
// slice each. Weight B-fragments pinned in VGPRs; A operands staged in LDS as
// bf16 in *fragment-swizzled* order so each lane's v16bf is one contiguous
// 32B ds load. Biases folded into WMMA C init. x staging double-buffered.
// All per-step addresses strength-reduced: per-thread slots precomputed once,
// pointers bumped by constant strides inside the T loop.
// ---------------------------------------------------------------------------

typedef __attribute__((ext_vector_type(16))) __bf16 v16bf;
typedef __attribute__((ext_vector_type(8)))  float  v8f;

#define B_  256
#define T_  2000
#define H_  64
#define G_  192
#define BM_ 16
#define NTHREADS 384   // 12 waves of 32

__device__ __forceinline__ unsigned short f2bf_u(float f) {
  unsigned u = __builtin_bit_cast(unsigned, f);
  unsigned r = u + 0x7FFFu + ((u >> 16) & 1u);   // round-to-nearest-even
  return (unsigned short)(r >> 16);
}
// Fast transcendentals: v_rcp_f32 instead of IEEE divide expansion.
__device__ __forceinline__ float sigmoid_fast(float x) {
  return __builtin_amdgcn_rcpf(1.0f + __expf(-x));
}
__device__ __forceinline__ float tanh_fast(float x) {
  return 1.0f - 2.0f * __builtin_amdgcn_rcpf(1.0f + __expf(2.0f * x));
}

// Fragment order within a 32-element K tile: identity (contiguous k = hi*16 + i)
// so lane half hi reads positions [hi*16, hi*16+16) as one contiguous v16bf.
__device__ __forceinline__ int swz8(int k) {
  int g  = (k >> 3) & 3;
  int gp = g;                                    // identity: 0->0, 1->1, 2->2, 3->3
  return (gp << 3) | (k & 7);
}
__device__ __forceinline__ int swzK(int k) {     // full K index (tile-preserving)
  return (k & ~31) | swz8(k & 31);
}

// ---------------------------------------------------------------------------
// Weight prep: fp32 -> bf16, pad W_ih0 K dim 80 -> 96 with zeros.
// ---------------------------------------------------------------------------
__global__ void prep_kernel(const float* __restrict__ Wih0, const float* __restrict__ Whh0,
                            const float* __restrict__ Wih1, const float* __restrict__ Whh1,
                            unsigned short* __restrict__ wih0b, unsigned short* __restrict__ whh0b,
                            unsigned short* __restrict__ wih1b, unsigned short* __restrict__ whh1b) {
  int i = blockIdx.x * blockDim.x + threadIdx.x;
  if (i < G_ * 96) {                       // W_ih0 padded [192][96]
    int g = i / 96, k = i - g * 96;
    wih0b[i] = (k < 80) ? f2bf_u(Wih0[g * 80 + k]) : (unsigned short)0;
  }
  int j = i - G_ * 96;
  if (j >= 0 && j < G_ * H_) whh0b[j] = f2bf_u(Whh0[j]);
  j -= G_ * H_;
  if (j >= 0 && j < G_ * H_) wih1b[j] = f2bf_u(Wih1[j]);
  j -= G_ * H_;
  if (j >= 0 && j < G_ * H_) whh1b[j] = f2bf_u(Whh1[j]);
}

// ---------------------------------------------------------------------------
// One GRU layer scan. KIN = padded input width (multiple of 32), FREAL = real
// input width in memory, KT_IN = KIN/32. IN_BF16: input tensor is bf16.
// ---------------------------------------------------------------------------
template <int KIN, int FREAL, int KT_IN, bool IN_BF16, bool WRITE_OUT>
__launch_bounds__(NTHREADS)
__global__ void gru_layer_kernel(const void* __restrict__ xin,
                                 const unsigned short* __restrict__ Wih,  // bf16 [192][KIN]
                                 const unsigned short* __restrict__ Whh,  // bf16 [192][64]
                                 const float* __restrict__ bih,
                                 const float* __restrict__ bhh,
                                 const float* __restrict__ h0,            // [256][64]
                                 float* __restrict__ hT,                  // [256][64]
                                 unsigned short* __restrict__ outbf) {    // bf16 [256][2000][64]
  __shared__ unsigned short xb[2][BM_ * KIN];  // staged input (bf16, swizzled, dbl-buf)
  __shared__ unsigned short hb[BM_ * H_];      // h as bf16 (swizzled) for WMMA A
  __shared__ float hs[BM_ * H_];               // h state fp32 for gate math
  __shared__ float gxa[G_ * BM_];              // gx pre-acts, [col][row] (transposed)
  __shared__ float gha[G_ * BM_];              // gh pre-acts, [col][row]

  const int tid  = threadIdx.x;
  const int wid  = tid >> 5;
  const int lane = tid & 31;
  const int b0   = blockIdx.x * BM_;
  const int m    = lane & 15;          // A-matrix row for this lane
  const int hi   = lane >> 4;          // half-wave selector
  const int n0   = wid * 16;           // gate-column base for this wave
  const int ncol = n0 + (lane & 15);   // C/D column for this lane
  constexpr int ESZ = IN_BF16 ? 2 : 4;
  constexpr int RB  = FREAL * ESZ;     // bytes per input row per timestep

  // ---- preload weight B-fragments into VGPRs (resident all T steps) -------
  // B layout: col N = lane&15; elements i cover K = kt*32 + hi*16 + i
  v16bf wihf[KT_IN];
  v16bf whhf[2];
  {
    const int g  = n0 + (lane & 15);
    const int kb = hi * 16;
#pragma unroll
    for (int kt = 0; kt < KT_IN; kt++)
      wihf[kt] = *(const v16bf*)(const void*)(Wih + g * KIN + kt * 32 + kb);
#pragma unroll
    for (int kt = 0; kt < 2; kt++)
      whhf[kt] = *(const v16bf*)(const void*)(Whh + g * H_ + kt * 32 + kb);
  }
  // Bias folded into WMMA C initializer (C/D rows share the lane's column).
  const float bi = bih[ncol];
  const float bh = bhh[ncol];

  // ---- precompute per-thread staging slots (addresses bump by RB per t) ---
  constexpr int HP   = FREAL / 2;                          // uint pairs per row
  constexpr int NSTG = (BM_ * HP + NTHREADS - 1) / NTHREADS;
  const char* ssrc[NSTG];
  int  sdst[NSTG];
  bool sact[NSTG];
#pragma unroll
  for (int s = 0; s < NSTG; s++) {
    int i = tid + s * NTHREADS;
    sact[s] = (i < BM_ * HP);
    int ii = sact[s] ? i : 0;
    int r = ii / HP, p = ii - r * HP, c = 2 * p;
    ssrc[s] = (const char*)xin + ((size_t)(b0 + r) * T_ * FREAL + c) * ESZ;
    sdst[s] = r * KIN + swzK(c);
  }
  auto do_stage = [&](unsigned short* dstbuf) {
#pragma unroll
    for (int s = 0; s < NSTG; s++) {
      bool on = (s == 0) ? true : sact[s];   // slot 0 provably full
      if (on) {
        unsigned u;
        if (IN_BF16) {
          u = *(const unsigned*)ssrc[s];
        } else {
          const float* p = (const float*)ssrc[s];
          u = (unsigned)f2bf_u(p[0]) | ((unsigned)f2bf_u(p[1]) << 16);
        }
        *(unsigned*)(dstbuf + sdst[s]) = u;
      }
      ssrc[s] += RB;                          // uniform advance (no guard)
    }
  };

  // ---- prefetch slot (one 64B line per thread, bumps by RB per t) ---------
  constexpr int LPR = (RB + 63) / 64;
  const bool pact = (tid < BM_ * LPR);
  const char* pf;
  {
    int ii = pact ? tid : 0;
    int r = ii / LPR, l = ii - r * LPR;
    pf = (const char*)xin + (size_t)(b0 + r) * T_ * RB + l * 64;
  }

  // ---- precompute per-thread gate slots -----------------------------------
  constexpr int NGT = (BM_ * H_ + NTHREADS - 1) / NTHREADS;  // 3
  int  g_i[NGT], g_rz[NGT], g_hb[NGT];
  bool g_act[NGT];
  unsigned short* g_out[NGT];
#pragma unroll
  for (int s = 0; s < NGT; s++) {
    int i = tid + s * NTHREADS;
    g_act[s] = (i < BM_ * H_);
    int ii = g_act[s] ? i : 0;
    int r = ii >> 6, j = ii & 63;
    g_i[s]  = ii;
    g_rz[s] = j * BM_ + r;                   // col-major base for r-gate column
    g_hb[s] = r * H_ + ((j & 32) | swz8(j & 31));
    g_out[s] = WRITE_OUT ? (outbf + (size_t)(b0 + r) * T_ * H_ + j) : (unsigned short*)0;
  }

  // ---- init LDS ------------------------------------------------------------
  for (int i = tid; i < 2 * BM_ * KIN; i += NTHREADS) (&xb[0][0])[i] = 0;  // pad stays 0
  for (int i = tid; i < BM_ * H_; i += NTHREADS) {
    int r = i >> 6, j = i & 63;
    float hv = h0[(size_t)(b0 + r) * H_ + j];
    hs[i] = hv;
    hb[r * H_ + ((j & 32) | swz8(j & 31))] = f2bf_u(hv);
  }
  __syncthreads();
  do_stage(&xb[0][0]);                       // stage t=0 (ssrc now at t=1)
  if (pact) __builtin_prefetch(pf + RB, 0, 1);   // prefetch t=1
  pf += 2 * (size_t)RB;                      // pf now at t=2
  __syncthreads();

  for (int t = 0; t < T_; t++) {
    const int cur = t & 1;
    // ---- stage x[t+1] into alternate buffer (overlaps WMMA below) ---------
    if (t + 1 < T_) do_stage(&xb[cur ^ 1][0]);
    if (t + 2 < T_) {
      if (pact) __builtin_prefetch(pf, 0, 1);
    }
    pf += RB;

    // ---- WMMA: gx = x_t @ Wih^T + b_ih, gh = h @ Whh^T + b_hh --------------
    const unsigned short* xrow = &xb[cur][m * KIN];
    const unsigned short* hrow = &hb[m * H_];
    v8f gx = {bi, bi, bi, bi, bi, bi, bi, bi};
    v8f gh = {bh, bh, bh, bh, bh, bh, bh, bh};
#pragma unroll
    for (int kt = 0; kt < KT_IN; kt++) {
      v16bf a = *(const v16bf*)(const void*)(xrow + kt * 32 + hi * 16);
      gx = __builtin_amdgcn_wmma_f32_16x16x32_bf16(false, a, false, wihf[kt],
                                                   (short)0, gx, false, false);
    }
#pragma unroll
    for (int kt = 0; kt < 2; kt++) {
      v16bf a = *(const v16bf*)(const void*)(hrow + kt * 32 + hi * 16);
      gh = __builtin_amdgcn_wmma_f32_16x16x32_bf16(false, a, false, whhf[kt],
                                                   (short)0, gh, false, false);
    }

    // ---- scatter pre-activations, branch-free -----------------------------
    // C/D layout: element j = (row hi*8+j, col ncol). Column-major LDS makes
    // each lane's 8 values contiguous -> one v8f store (2x ds_store_b128).
    *(v8f*)(&gxa[ncol * BM_ + hi * 8]) = gx;
    *(v8f*)(&gha[ncol * BM_ + hi * 8]) = gh;
    __syncthreads();

    // ---- gate math + state update (slot-unrolled) -------------------------
#pragma unroll
    for (int s = 0; s < NGT; s++) {
      bool on = (s + 1 < NGT) ? true : g_act[s];  // slots 0..NGT-2 provably full
      if (on) {
        const int base = g_rz[s];
        float rg = sigmoid_fast(gxa[base] + gha[base]);
        float zg = sigmoid_fast(gxa[base + 64 * BM_] + gha[base + 64 * BM_]);
        float ng = tanh_fast(gxa[base + 128 * BM_] + rg * gha[base + 128 * BM_]);
        float hv = (1.0f - zg) * ng + zg * hs[g_i[s]];
        hs[g_i[s]] = hv;
        unsigned short b16 = f2bf_u(hv);
        hb[g_hb[s]] = b16;
        if (WRITE_OUT) {
          *g_out[s] = b16;
          g_out[s] += H_;
        }
      }
    }
    __syncthreads();
  }

  // ---- write final hidden state -------------------------------------------
  for (int i = tid; i < BM_ * H_; i += NTHREADS)
    hT[(size_t)(b0 + (i >> 6)) * H_ + (i & 63)] = hs[i];
}

// ---------------------------------------------------------------------------
// logits = relu(h1T) @ W_fc^T + b_fc   (tiny: 256x29x64)
// ---------------------------------------------------------------------------
__global__ void fc_kernel(const float* __restrict__ h1T, const float* __restrict__ Wfc,
                          const float* __restrict__ bfc, float* __restrict__ logits) {
  int b = blockIdx.x;
  int o = threadIdx.x;
  if (o < 29) {
    float acc = bfc[o];
#pragma unroll
    for (int j = 0; j < H_; j++) {
      float hv = h1T[b * H_ + j];
      hv = hv > 0.0f ? hv : 0.0f;
      acc += hv * Wfc[o * H_ + j];
    }
    logits[b * 29 + o] = acc;
  }
}

// ---------------------------------------------------------------------------
extern "C" void kernel_launch(void* const* d_in, const int* in_sizes, int n_in,
                              void* d_out, int out_size, void* d_ws, size_t ws_size,
                              hipStream_t stream) {
  (void)in_sizes; (void)n_in; (void)out_size; (void)ws_size;

  const float* x     = (const float*)d_in[0];   // [256,2000,80]
  const float* h     = (const float*)d_in[1];   // [2,256,64]
  const float* W_ih0 = (const float*)d_in[2];   // [192,80]
  const float* W_hh0 = (const float*)d_in[3];   // [192,64]
  const float* b_ih0 = (const float*)d_in[4];
  const float* b_hh0 = (const float*)d_in[5];
  const float* W_ih1 = (const float*)d_in[6];   // [192,64]
  const float* W_hh1 = (const float*)d_in[7];   // [192,64]
  const float* b_ih1 = (const float*)d_in[8];
  const float* b_hh1 = (const float*)d_in[9];
  const float* W_fc  = (const float*)d_in[10];  // [29,64]
  const float* b_fc  = (const float*)d_in[11];

  float* out    = (float*)d_out;
  float* logits = out;                          // [256,29]
  float* hid    = out + 256 * 29;               // [2,256,64]

  char* ws = (char*)d_ws;
  unsigned short* wih0b = (unsigned short*)(ws + 0);        // 192*96*2  = 36864
  unsigned short* whh0b = (unsigned short*)(ws + 36864);    // 24576
  unsigned short* wih1b = (unsigned short*)(ws + 61440);    // 24576
  unsigned short* whh1b = (unsigned short*)(ws + 86016);    // 24576
  unsigned short* out0b = (unsigned short*)(ws + 110592);   // 256*2000*64*2 = 65.5 MB

  // 1) weights -> bf16 (padded)
  prep_kernel<<<(G_ * 96 + 3 * G_ * H_ + 255) / 256, 256, 0, stream>>>(
      W_ih0, W_hh0, W_ih1, W_hh1, wih0b, whh0b, wih1b, whh1b);

  // 2) layer 0 scan (input fp32, K=80 padded to 96, writes out0 as bf16)
  gru_layer_kernel<96, 80, 3, false, true><<<B_ / BM_, NTHREADS, 0, stream>>>(
      (const void*)x, wih0b, whh0b, b_ih0, b_hh0,
      h /* layer 0 h0 */, hid /* h0T */, out0b);

  // 3) layer 1 scan (input bf16, K=64; only final h needed)
  gru_layer_kernel<64, 64, 2, true, false><<<B_ / BM_, NTHREADS, 0, stream>>>(
      (const void*)out0b, wih1b, whh1b, b_ih1, b_hh1,
      h + 256 * 64 /* layer 1 h0 */, hid + 256 * 64 /* h1T */, nullptr);

  // 4) fc(relu(h1T))
  fc_kernel<<<B_, 32, 0, stream>>>(hid + 256 * 64, W_fc, b_fc, logits);
}
